// SelfAttentionEnhanced_65481071395381
// MI455X (gfx1250) — matrix operand order repaired and checked by reading, not compiled, not run on hardware
//
#include <hip/hip_runtime.h>
#include <hip/hip_bf16.h>
#include <stdint.h>

#define BATCH 4
#define SEQ   4096
#define DIM   512
#define BS    (BATCH * SEQ)

typedef __attribute__((ext_vector_type(16))) __bf16 v16bf;
typedef __attribute__((ext_vector_type(8)))  __bf16 bf16x8;
typedef __attribute__((ext_vector_type(4)))  __bf16 bf16x4;
typedef __attribute__((ext_vector_type(8)))  float  v8f;

#define SHUF16(lo, hi) __builtin_shufflevector((lo), (hi), 0,1,2,3,4,5,6,7,8,9,10,11,12,13,14,15)

// Scheduler pipeline pinning (composable-kernel style).
// 0x020 = VMEM read group, 0x008 = MFMA/WMMA group.
#if __has_builtin(__builtin_amdgcn_sched_group_barrier)
#define SGB(mask, cnt) __builtin_amdgcn_sched_group_barrier((mask), (cnt), 0)
#else
#define SGB(mask, cnt)
#endif

// Pin "8 loads; 12x(1 wmma, 2 loads); 4 wmma" onto a 32-load/16-WMMA block.
__device__ __forceinline__ void pipeline_32load_16wmma() {
  SGB(0x020, 8);
#pragma unroll
  for (int i = 0; i < 12; ++i) { SGB(0x008, 1); SGB(0x020, 2); }
  SGB(0x008, 4);
}

__device__ __forceinline__ __bf16 f2bf(float f) {
  union { float f; uint32_t u; } x; x.f = f;
  uint32_t r = x.u + 0x7FFFu + ((x.u >> 16) & 1u);  // round-to-nearest-even
  uint16_t h = (uint16_t)(r >> 16);
  return __builtin_bit_cast(__bf16, h);
}

// ---------------- fp32 -> bf16 conversion ----------------
__global__ void cvt_f32_bf16(const float* __restrict__ src, __bf16* __restrict__ dst, int n) {
  int i = (blockIdx.x * blockDim.x + threadIdx.x) * 4;
  if (i < n) {
    float4 v = *(const float4*)(src + i);
    bf16x4 o;
    o[0] = f2bf(v.x); o[1] = f2bf(v.y); o[2] = f2bf(v.z); o[3] = f2bf(v.w);
    *(bf16x4*)(dst + i) = o;
  }
}

// ---------------- Q/K/V projection: C = X * W^T ----------------
__global__ __launch_bounds__(128) __attribute__((amdgpu_waves_per_eu(2)))
void proj_kernel(const __bf16* __restrict__ X,
                 const __bf16* __restrict__ Wall,
                 __bf16* __restrict__ Qb,   // [BS][DIM]
                 __bf16* __restrict__ Kb,   // [BS][DIM]
                 __bf16* __restrict__ Vtb,  // [BATCH][DIM][SEQ]
                 const float* __restrict__ temp)
{
  const int tid  = threadIdx.x;
  const int lane = tid & 31;
  const int wave = tid >> 5;
  const int lrow = lane & 15;
  const int half = (lane >> 4) & 1;

  const int sTile = blockIdx.x & 1023;   // 1024 s-tiles of 16 rows
  const int m     = blockIdx.x >> 10;    // 0=Q 1=K 2=V
  const int sBase = sTile * 16;

  const __bf16* W = Wall + (size_t)m * DIM * DIM;

  // Preload A tiles: lane holds row (sBase+lrow), 16 chunks of K=32.
  const __bf16* xrow = X + (size_t)(sBase + lrow) * DIM;
  v16bf a[16];
#pragma unroll
  for (int ec = 0; ec < 16; ++ec) {
    const int d0 = ec * 32 + 8 * half;
    bf16x8 lo = *(const bf16x8*)(xrow + d0);
    bf16x8 hi = *(const bf16x8*)(xrow + d0 + 16);
    a[ec] = SHUF16(lo, hi);
  }

  const float alpha = temp[0] * 0.04419417382415922f;  // 1/sqrt(512)

  for (int et = 0; et < 8; ++et) {
    const int eBase = (wave * 8 + et) * 16;
    const int ecol  = eBase + lrow;                // lane's output column e
    const __bf16* wrow = W + (size_t)ecol * DIM;

    v16bf bch[16];
#pragma unroll
    for (int ec = 0; ec < 16; ++ec) {
      const int d0 = ec * 32 + 8 * half;
      bf16x8 blo = *(const bf16x8*)(wrow + d0);
      bf16x8 bhi = *(const bf16x8*)(wrow + d0 + 16);
      bch[ec] = SHUF16(blo, bhi);
    }
    // Two independent accumulator chains, summed at the end (fp32, exact).
    v8f c0 = {0.f,0.f,0.f,0.f,0.f,0.f,0.f,0.f};
    v8f c1 = {0.f,0.f,0.f,0.f,0.f,0.f,0.f,0.f};
#pragma unroll
    for (int ec = 0; ec < 8; ++ec) {
      c0 = __builtin_amdgcn_wmma_f32_16x16x32_bf16(false, a[2*ec],   false, bch[2*ec],
                                                   (short)0, c0, false, false);
      c1 = __builtin_amdgcn_wmma_f32_16x16x32_bf16(false, a[2*ec+1], false, bch[2*ec+1],
                                                   (short)0, c1, false, false);
    }
    pipeline_32load_16wmma();   // pin 4-chunk-deep load/WMMA software pipeline
    v8f c = c0 + c1;

    if (m == 0) {
#pragma unroll
      for (int v = 0; v < 8; ++v)
        Qb[(size_t)(sBase + v + 8 * half) * DIM + ecol] = f2bf(c[v] * alpha);
    } else if (m == 1) {
#pragma unroll
      for (int v = 0; v < 8; ++v)
        Kb[(size_t)(sBase + v + 8 * half) * DIM + ecol] = f2bf(c[v]);
    } else {
      const int b_  = sBase >> 12;                 // batch
      const int sl  = (sBase & 4095) + 8 * half;   // 8 consecutive s values
      bf16x8 o;
#pragma unroll
      for (int v = 0; v < 8; ++v) o[v] = f2bf(c[v]);
      *(bf16x8*)(Vtb + ((size_t)b_ * DIM + ecol) * SEQ + sl) = o;
    }
  }
}

// ---------------- flash attention ----------------
// One 16-row q tile per block (128 threads = 4 waves).
__global__ __launch_bounds__(128) __attribute__((amdgpu_waves_per_eu(2)))
void attn_kernel(const __bf16* __restrict__ Qb,
                 const __bf16* __restrict__ Kb,
                 const __bf16* __restrict__ Vtb,
                 float* __restrict__ Out)
{
  __shared__ __align__(16) float  sS[16][64];
  __shared__ __align__(16) __bf16 sP[16][64];
  __shared__ float sRed[16][8];   // row-max partials
  __shared__ float sSum[16][8];   // row-sum partials
  __shared__ float sCorr[16];
  __shared__ float sM[16];
  __shared__ float sL[16];

  const int tid  = threadIdx.x;
  const int lane = tid & 31;
  const int wave = tid >> 5;
  const int lrow = lane & 15;
  const int half = (lane >> 4) & 1;

  const int qBase = blockIdx.x * 16;   // global row in [0, BS)
  const int bidx  = qBase >> 12;       // batch
  const int kRow0 = bidx * SEQ;

  // Preload Q A-tiles (16 chunks x K=32) — reused every iteration.
  const __bf16* qrow = Qb + (size_t)(qBase + lrow) * DIM;
  v16bf qa[16];
#pragma unroll
  for (int ec = 0; ec < 16; ++ec) {
    const int d0 = ec * 32 + 8 * half;
    bf16x8 lo = *(const bf16x8*)(qrow + d0);
    bf16x8 hi = *(const bf16x8*)(qrow + d0 + 16);
    qa[ec] = SHUF16(lo, hi);
  }

  v8f o[8];
#pragma unroll
  for (int i = 0; i < 8; ++i) o[i] = (v8f){0.f,0.f,0.f,0.f,0.f,0.f,0.f,0.f};

  if (tid < 16) { sM[tid] = -3.0e38f; sL[tid] = 0.0f; }
  __syncthreads();

  const int r = tid >> 3;   // softmax row this thread owns a slice of
  const int g = tid & 7;    // 8-column group within the 64-col chunk

  for (int it = 0; it < SEQ / 64; ++it) {
    const int kChunk = it * 64;

    // ---- scores: this wave's 16 k-columns, full 512-deep accumulate ----
    const int kcol = kChunk + wave * 16 + lrow;
    const __bf16* krow = Kb + (size_t)(kRow0 + kcol) * DIM;

    v16bf kch[16];
#pragma unroll
    for (int ec = 0; ec < 16; ++ec) {
      const int d0 = ec * 32 + 8 * half;
      bf16x8 blo = *(const bf16x8*)(krow + d0);
      bf16x8 bhi = *(const bf16x8*)(krow + d0 + 16);
      kch[ec] = SHUF16(blo, bhi);
    }
    v8f c0 = {0.f,0.f,0.f,0.f,0.f,0.f,0.f,0.f};
    v8f c1 = {0.f,0.f,0.f,0.f,0.f,0.f,0.f,0.f};
#pragma unroll
    for (int ec = 0; ec < 8; ++ec) {
      c0 = __builtin_amdgcn_wmma_f32_16x16x32_bf16(false, qa[2*ec],   false, kch[2*ec],
                                                   (short)0, c0, false, false);
      c1 = __builtin_amdgcn_wmma_f32_16x16x32_bf16(false, qa[2*ec+1], false, kch[2*ec+1],
                                                   (short)0, c1, false, false);
    }
    pipeline_32load_16wmma();   // pin 4-chunk-deep load/WMMA software pipeline
    v8f c = c0 + c1;

#pragma unroll
    for (int v = 0; v < 8; ++v)
      sS[v + 8 * half][wave * 16 + lrow] = c[v];
    __syncthreads();                                   // #1 scores visible

    // ---- online softmax over the 64-col chunk ----
    float sv[8];
    float mloc = -3.0e38f;
#pragma unroll
    for (int j = 0; j < 8; ++j) { sv[j] = sS[r][g * 8 + j]; mloc = fmaxf(mloc, sv[j]); }
    sRed[r][g] = mloc;
    __syncthreads();                                   // #2 partial maxes visible
    float mchunk = sRed[r][0];
#pragma unroll
    for (int j = 1; j < 8; ++j) mchunk = fmaxf(mchunk, sRed[r][j]);
    const float mold = sM[r];
    const float mnew = fmaxf(mold, mchunk);
    const float corr = __expf(mold - mnew);
    float lpart = 0.f;
#pragma unroll
    for (int j = 0; j < 8; ++j) {
      float p = __expf(sv[j] - mnew);
      lpart += p;
      sP[r][g * 8 + j] = f2bf(p);
    }
    sSum[r][g] = lpart;
    if (g == 0) sCorr[r] = corr;
    __syncthreads();                                   // #3 P, sums, corr visible

    if (g == 0) {
      float l = 0.f;
#pragma unroll
      for (int j = 0; j < 8; ++j) l += sSum[r][j];
      sL[r] = sL[r] * corr + l;
      sM[r] = mnew;
    }

    // ---- O = O*corr + P·V (this wave owns e = wave*128 .. +127) ----
    float corrv[8];
#pragma unroll
    for (int v = 0; v < 8; ++v) corrv[v] = sCorr[v + 8 * half];
#pragma unroll
    for (int nt = 0; nt < 8; ++nt)
#pragma unroll
      for (int v = 0; v < 8; ++v) o[nt][v] *= corrv[v];

#pragma unroll
    for (int t = 0; t < 2; ++t) {
      bf16x8 plo = *(const bf16x8*)(&sP[lrow][t * 32 + 8 * half]);
      bf16x8 phi = *(const bf16x8*)(&sP[lrow][t * 32 + 16 + 8 * half]);
      v16bf pa = SHUF16(plo, phi);

      // Double-buffered V tile loads: load nt+1 while WMMA nt runs.
      v16bf vb[2];
      {
        const int e0 = wave * 128 + lrow;
        const __bf16* vrow0 = Vtb + ((size_t)bidx * DIM + e0) * SEQ
                            + kChunk + t * 32 + 8 * half;
        bf16x8 vlo = *(const bf16x8*)(vrow0);
        bf16x8 vhi = *(const bf16x8*)(vrow0 + 16);
        vb[0] = SHUF16(vlo, vhi);
      }
#pragma unroll
      for (int nt = 0; nt < 8; ++nt) {
        if (nt + 1 < 8) {
          const int e1 = wave * 128 + (nt + 1) * 16 + lrow;
          const __bf16* vrow1 = Vtb + ((size_t)bidx * DIM + e1) * SEQ
                              + kChunk + t * 32 + 8 * half;
          bf16x8 vlo = *(const bf16x8*)(vrow1);
          bf16x8 vhi = *(const bf16x8*)(vrow1 + 16);
          if (it + 1 < SEQ / 64) __builtin_prefetch(vrow1 + 64, 0, 3);
          vb[(nt + 1) & 1] = SHUF16(vlo, vhi);
        }
        o[nt] = __builtin_amdgcn_wmma_f32_16x16x32_bf16(false, pa, false, vb[nt & 1],
                                                        (short)0, o[nt], false, false);
      }
    }
    __syncthreads();                                   // #4 end of iteration
  }

  // ---- epilogue: divide by running sum, write fp32 ----
  float lv[8];
#pragma unroll
  for (int v = 0; v < 8; ++v) lv[v] = 1.0f / sL[v + 8 * half];
#pragma unroll
  for (int nt = 0; nt < 8; ++nt) {
    const int e = wave * 128 + nt * 16 + lrow;
#pragma unroll
    for (int v = 0; v < 8; ++v)
      Out[(size_t)(qBase + v + 8 * half) * DIM + e] = o[nt][v] * lv[v];
  }
}

extern "C" void kernel_launch(void* const* d_in, const int* in_sizes, int n_in,
                              void* d_out, int out_size, void* d_ws, size_t ws_size,
                              hipStream_t stream) {
  const float* X    = (const float*)d_in[0];
  const float* Wq   = (const float*)d_in[1];
  const float* Wk   = (const float*)d_in[2];
  const float* Wv   = (const float*)d_in[3];
  const float* temp = (const float*)d_in[4];
  float* Out = (float*)d_out;

  // Workspace carve-up (all 256B aligned). Total ~66 MB.
  uint8_t* ws = (uint8_t*)d_ws;
  size_t off = 0;
  auto carve = [&](size_t bytes) -> void* {
    void* p = ws + off;
    off += (bytes + 255) & ~(size_t)255;
    return p;
  };
  __bf16* Xb  = (__bf16*)carve((size_t)BS * DIM * 2);
  __bf16* Wb  = (__bf16*)carve((size_t)3 * DIM * DIM * 2);
  __bf16* Qb  = (__bf16*)carve((size_t)BS * DIM * 2);
  __bf16* Kb  = (__bf16*)carve((size_t)BS * DIM * 2);
  __bf16* Vtb = (__bf16*)carve((size_t)BATCH * DIM * SEQ * 2);

  // 1) fp32 -> bf16
  {
    int n = BS * DIM;
    cvt_f32_bf16<<<n / 4 / 256, 256, 0, stream>>>(X, Xb, n);
    int nw = DIM * DIM;
    cvt_f32_bf16<<<nw / 4 / 256, 256, 0, stream>>>(Wq, Wb + 0 * (size_t)DIM * DIM, nw);
    cvt_f32_bf16<<<nw / 4 / 256, 256, 0, stream>>>(Wk, Wb + 1 * (size_t)DIM * DIM, nw);
    cvt_f32_bf16<<<nw / 4 / 256, 256, 0, stream>>>(Wv, Wb + 2 * (size_t)DIM * DIM, nw);
  }

  // 2) Q/K/V projections (Q pre-scaled, V transposed)
  proj_kernel<<<3 * (BS / 16), 128, 0, stream>>>(Xb, Wb, Qb, Kb, Vtb, temp);

  // 3) flash attention
  attn_kernel<<<BS / 16, 128, 0, stream>>>(Qb, Kb, Vtb, Out);
}
